// MoETransformerBlock_2224793060080
// MI455X (gfx1250) — matrix operand order repaired and checked by reading, not compiled, not run on hardware
//
#include <hip/hip_runtime.h>
#include <hip/hip_bf16.h>
#include <math.h>

typedef _Float16 half_t;
typedef __attribute__((ext_vector_type(16))) _Float16 v16h;
typedef __attribute__((ext_vector_type(8)))  _Float16 v8h;
typedef __attribute__((ext_vector_type(8)))  float    v8f;
typedef __attribute__((ext_vector_type(4)))  unsigned v4u;
typedef __attribute__((ext_vector_type(8)))  int      v8i;
typedef __attribute__((ext_vector_type(4)))  int      v4i;

// ---------------- problem constants ----------------
constexpr int cB   = 8;
constexpr int cN   = 1024;
constexpr int cC   = 1152;
constexpr int cH   = 4608;     // 4*C
constexpr int cE   = 4;
constexpr int cNH  = 16;
constexpr int cHD  = 72;       // C / NH
constexpr int cHDP = 96;       // padded head dim (multiple of 32)
constexpr int cT   = cB * cN;  // 8192 tokens
constexpr int c3C  = 3 * cC;
constexpr int c6C  = 6 * cC;
constexpr int cBH  = cB * cNH;

__device__ inline float gelu_tanh(float x) {
    float x3 = x * x * x;
    float t  = tanhf(0.7978845608028654f * (x + 0.044715f * x3));
    return 0.5f * x * (1.f + t);
}

// ==========================================================================
// Generic batched WMMA GEMM:  C = epi(alpha * A[f16] x B[f16] + bias)
//   A: [M,K] row-major (lda), B: [K,N] row-major (ldb), batch via blockIdx.z
//   Block = 256 threads (8 waves, 2x4), macro tile 64x128, K step 64
//   (2 WMMA K-slices per stage => 8 v_wmma per barrier pair).
//   FULL: M%64==0 && N%128==0 && K%64==0 -> no bounds checks.
//   TDM : stage the A tile with the Tensor Data Mover (wave 0 issues the DMA,
//         LDS row padding produced by the D# pad feature), drained with
//         s_wait_tensorcnt.
// epi: 0=f32 store  1=f16 store  2=f16 gelu  3=f32 gated residual
//      4=f32 moe accumulate (Cout += gate[row]*val, overwrite if accumFirst)
// ==========================================================================
template <bool FULL, bool TDM>
__global__ __launch_bounds__(256) void wmma_gemm_kernel(
    const half_t* __restrict__ A, const half_t* __restrict__ Bm,
    const float* __restrict__ bias, void* __restrict__ Cout,
    int M, int N, int K, int lda, int ldb, int ldc,
    long long sA, long long sB, long long sC,
    float alpha, int epi,
    const float* __restrict__ resid,
    const float* __restrict__ gate, long long gateLd, int rowsPerBatch,
    int accumFirst)
{
    constexpr int BM = 64, BN = 128, BK = 64;
    constexpr int AKP  = BK + 8;      // 72 halves/row = 144B (16B aligned)
    constexpr int BKPW = BK / 2 + 4;  // 36 words/row  = 144B (16B aligned)
    __shared__ half_t   As[BM][AKP]  __attribute__((aligned(16)));
    __shared__ unsigned Bs[BN][BKPW] __attribute__((aligned(16)));

    const int tid  = threadIdx.x;
    const int wave = tid >> 5;
    const int lane = tid & 31;
    const int wm   = wave >> 2;      // 0..1
    const int wn   = wave & 3;       // 0..3
    const int l16  = lane >> 4;      // 0/1 (half-wave)
    const int lm   = lane & 15;

    const long long zb = blockIdx.z;
    const half_t* Ab = A  + zb * sA;
    const half_t* Bb = Bm + zb * sB;
    const int tileM = blockIdx.y * BM;
    const int tileN = blockIdx.x * BN;

    v8f acc[2][2];
    #pragma unroll
    for (int i = 0; i < 2; ++i)
        #pragma unroll
        for (int j = 0; j < 2; ++j)
            acc[i][j] = v8f{};

    // cooperative staging indices
    const int arow = tid >> 2;        // 0..63   (A rows)
    const int akc  = (tid & 3) * 16;  // 0,16,32,48
    const int bkp  = tid >> 3;        // 0..31   (B k-pairs)
    const int bnc  = (tid & 7) * 16;  // 0..112

    for (int k0 = 0; k0 < K; k0 += BK) {
        // ---------------- stage A tile ----------------
        if (TDM) {
            if (wave == 0) {
                // Tensor DMA: 2D tile 64(K) x 64(M) halves, row stride = lda,
                // LDS pad: every 32 DWORDs (=64 halves) insert 4 DWORDs
                // (=8 halves) -> 72-half LDS rows matching As[][AKP].
                const unsigned long long ga =
                    (unsigned long long)(uintptr_t)(Ab + (long long)tileM * lda + k0);
                const unsigned ldsA = (unsigned)(uintptr_t)&As[0][0];
                v4u g0;
                g0[0] = 1u;                                   // count=1 (valid user D#)
                g0[1] = ldsA;                                 // lds_addr
                g0[2] = (unsigned)(ga & 0xFFFFFFFFu);         // global_addr[31:0]
                g0[3] = (unsigned)((ga >> 32) & 0x01FFFFFFu)  // global_addr[56:32]
                        | (2u << 30);                         // type = 2 (image)
                v8i g1;
                g1[0] = (int)((1u << 16)      // data_size = 2B
                        | (1u << 20)          // pad_enable
                        | (4u << 22)          // pad_interval: code 4 = 32 DWORDs
                        | (3u << 25));        // pad_amount:   code 3 = 4 DWORDs
                g1[1] = (int)(((unsigned)K & 0xFFFFu) << 16);                 // tensor_dim0 lo
                g1[2] = (int)((((unsigned)K >> 16) & 0xFFFFu)
                        | (((unsigned)M & 0xFFFFu) << 16));                   // dim0 hi | dim1 lo
                g1[3] = (int)((((unsigned)M >> 16) & 0xFFFFu) | (64u << 16)); // dim1 hi | tile_dim0
                g1[4] = (int)64u;                                             // tile_dim1 (tile_dim2=0)
                g1[5] = (int)(unsigned)lda;                                   // tensor_dim0_stride lo
                g1[6] = 0;                                                    // stride hi | dim1_stride lo
                g1[7] = 0;
                v4i z4 = {};
#if __clang_major__ >= 23
                v8i z8 = {};
                __builtin_amdgcn_tensor_load_to_lds(g0, g1, z4, z4, z8, 0);
#else
                __builtin_amdgcn_tensor_load_to_lds(g0, g1, z4, z4, 0);
#endif
                __builtin_amdgcn_s_wait_tensorcnt(0);
            }
        } else {
            const int gr = tileM + arow;
            const half_t* ap = Ab + (long long)gr * lda + (k0 + akc);
            v8h a0 = {}, a1 = {};
            if (FULL || (gr < M && k0 + akc     < K)) a0 = *(const v8h*)ap;
            if (FULL || (gr < M && k0 + akc + 8 < K)) a1 = *(const v8h*)(ap + 8);
            *(v8h*)&As[arow][akc]     = a0;
            *(v8h*)&As[arow][akc + 8] = a1;
        }
        // ---------------- stage B tile, transposed, K-pair packed ----------
        {
            const int r0 = k0 + 2 * bkp;
            const half_t* p0 = Bb + (long long)r0 * ldb + tileN + bnc;
            const half_t* p1 = p0 + ldb;
            v8h x0 = {}, x1 = {}, x2 = {}, x3 = {};
            if (FULL || (r0     < K && tileN + bnc     < N)) x0 = *(const v8h*)p0;
            if (FULL || (r0     < K && tileN + bnc + 8 < N)) x1 = *(const v8h*)(p0 + 8);
            if (FULL || (r0 + 1 < K && tileN + bnc     < N)) x2 = *(const v8h*)p1;
            if (FULL || (r0 + 1 < K && tileN + bnc + 8 < N)) x3 = *(const v8h*)(p1 + 8);
            union HP { half_t h[2]; unsigned u; } pk;
            #pragma unroll
            for (int i = 0; i < 8; ++i) {
                pk.h[0] = x0[i]; pk.h[1] = x2[i];
                Bs[bnc + i][bkp] = pk.u;
            }
            #pragma unroll
            for (int i = 0; i < 8; ++i) {
                pk.h[0] = x1[i]; pk.h[1] = x3[i];
                Bs[bnc + 8 + i][bkp] = pk.u;
            }
            if (k0 + BK < K)
                __builtin_prefetch((const void*)(p0 + (long long)BK * ldb), 0, 0);
        }
        __syncthreads();

        // -------- 2 K-slices x 4 WMMA, fragments in contiguous k order (16h+s) ----
        union Frag { v16h v; v8h h[2]; };
        #pragma unroll
        for (int ks = 0; ks < BK; ks += 32) {
            Frag a[2], bf[2];
            #pragma unroll
            for (int mt = 0; mt < 2; ++mt) {
                const int r = wm * 32 + mt * 16 + lm;
                // lanes 0-15: K ks..ks+15 ; lanes 16-31: K ks+16..ks+31
                a[mt].h[0] = *(const v8h*)&As[r][ks + l16 * 16];
                a[mt].h[1] = *(const v8h*)&As[r][ks + l16 * 16 + 8];
            }
            #pragma unroll
            for (int nt = 0; nt < 2; ++nt) {
                const int cq = wn * 32 + nt * 16 + lm;
                const half_t* bbase = (const half_t*)&Bs[cq][0];
                // lanes 0-15: K ks..ks+15 ; lanes 16-31: K ks+16..ks+31
                bf[nt].h[0] = *(const v8h*)(bbase + ks + l16 * 16);
                bf[nt].h[1] = *(const v8h*)(bbase + ks + l16 * 16 + 8);
            }
            #pragma unroll
            for (int mt = 0; mt < 2; ++mt)
                #pragma unroll
                for (int nt = 0; nt < 2; ++nt)
                    acc[mt][nt] = __builtin_amdgcn_wmma_f32_16x16x32_f16(
                        false, a[mt].v, false, bf[nt].v,
                        (short)0, acc[mt][nt], false, false);
        }
        __syncthreads();
    }

    // ---- epilogue / store (C/D layout: vgpr v -> M=v / v+8, lanes -> N) ----
    #pragma unroll
    for (int mt = 0; mt < 2; ++mt)
    #pragma unroll
    for (int nt = 0; nt < 2; ++nt)
    #pragma unroll
    for (int v = 0; v < 8; ++v) {
        const int row = tileM + wm * 32 + mt * 16 + l16 * 8 + v;
        const int col = tileN + wn * 32 + nt * 16 + lm;
        if (!FULL && (row >= M || col >= N)) continue;
        float val = alpha * acc[mt][nt][v] + (bias ? bias[col] : 0.f);
        const long long ci = zb * sC + (long long)row * ldc + col;
        if (epi == 0) {
            ((float*)Cout)[ci] = val;
        } else if (epi == 1) {
            ((half_t*)Cout)[ci] = (half_t)val;
        } else if (epi == 2) {
            ((half_t*)Cout)[ci] = (half_t)gelu_tanh(val);
        } else if (epi == 3) {
            const int b = row / rowsPerBatch;
            ((float*)Cout)[ci] = resid[ci] + gate[(long long)b * gateLd + col] * val;
        } else { // epi == 4
            const float w = gate[(long long)row * gateLd];
            const float prev = accumFirst ? 0.f : ((float*)Cout)[ci];
            ((float*)Cout)[ci] = prev + w * val;
        }
    }
}

// ==========================================================================
// small helper kernels
// ==========================================================================
__global__ void silu_kernel(const float* __restrict__ c, float* __restrict__ o, int n) {
    int i = blockIdx.x * blockDim.x + threadIdx.x;
    if (i < n) { float v = c[i]; o[i] = v / (1.f + expf(-v)); }
}

__global__ void f32_to_f16_kernel(const float* __restrict__ s, half_t* __restrict__ d, long long n) {
    long long i = (long long)blockIdx.x * blockDim.x + threadIdx.x;
    if (i < n) d[i] = (half_t)s[i];
}

// mod[b, col] = silu(c)[b,:] dot w_adaln[:, col] + b_adaln[col]
__global__ void adaln_kernel(const float* __restrict__ cs, const float* __restrict__ w,
                             const float* __restrict__ bias, float* __restrict__ mod) {
    int idx = blockIdx.x * blockDim.x + threadIdx.x;
    if (idx >= cB * c6C) return;
    int b = idx / c6C, col = idx % c6C;
    float acc = bias[col];
    const float* cr = cs + (long long)b * cC;
    for (int k = 0; k < cC; ++k) acc += cr[k] * w[(long long)k * c6C + col];
    mod[idx] = acc;
}

// LayerNorm (no affine) + modulate -> f16
__global__ __launch_bounds__(256) void ln_mod_kernel(
    const float* __restrict__ X, const float* __restrict__ mod,
    int shiftOff, int scaleOff, half_t* __restrict__ out)
{
    const int t = blockIdx.x;
    const int b = t / cN;
    const float* x = X + (long long)t * cC;
    __shared__ float r1[256], r2[256];
    float s = 0.f, s2 = 0.f;
    for (int c = threadIdx.x; c < cC; c += 256) { float v = x[c]; s += v; s2 += v * v; }
    r1[threadIdx.x] = s; r2[threadIdx.x] = s2;
    __syncthreads();
    for (int k = 128; k > 0; k >>= 1) {
        if (threadIdx.x < k) { r1[threadIdx.x] += r1[threadIdx.x + k]; r2[threadIdx.x] += r2[threadIdx.x + k]; }
        __syncthreads();
    }
    const float mu  = r1[0] / cC;
    const float var = r2[0] / cC - mu * mu;
    const float rs  = rsqrtf(var + 1e-6f);
    const float* sh = mod + (long long)b * c6C + shiftOff;
    const float* sc = mod + (long long)b * c6C + scaleOff;
    half_t* o = out + (long long)t * cC;
    for (int c = threadIdx.x; c < cC; c += 256)
        o[c] = (half_t)(((x[c] - mu) * rs) * (1.f + sc[c]) + sh[c]);
}

// split qkv (f16 [T,3C]) into padded Q [bh,n,96], K^T [bh,96,n], V [bh,n,96]
__global__ void qkv_split_kernel(const half_t* __restrict__ qkv,
                                 half_t* __restrict__ Q, half_t* __restrict__ Kt,
                                 half_t* __restrict__ V)
{
    long long idx = (long long)blockIdx.x * blockDim.x + threadIdx.x;
    const long long total = (long long)cBH * cN * cHDP;
    if (idx >= total) return;
    int d  = (int)(idx % cHDP);
    int n  = (int)((idx / cHDP) % cN);
    int bh = (int)(idx / ((long long)cHDP * cN));
    int b = bh / cNH, h = bh % cNH;
    long long t = (long long)b * cN + n;
    half_t qv = (half_t)0.f, kv = (half_t)0.f, vv = (half_t)0.f;
    if (d < cHD) {
        const half_t* base = qkv + t * c3C + h * cHD + d;
        qv = base[0]; kv = base[cC]; vv = base[2 * cC];
    }
    Q [(long long)bh * cN * cHDP + (long long)n * cHDP + d] = qv;
    Kt[(long long)bh * cHDP * cN + (long long)d * cN + n]   = kv;
    V [(long long)bh * cN * cHDP + (long long)n * cHDP + d] = vv;
}

__global__ __launch_bounds__(256) void softmax_kernel(
    const float* __restrict__ S, half_t* __restrict__ P)
{
    const long long row = (long long)blockIdx.y * gridDim.x + blockIdx.x;
    const float* s = S + row * cN;
    half_t* p = P + row * cN;
    __shared__ float red[256];
    float m = -3.4e38f;
    for (int c = threadIdx.x; c < cN; c += 256) m = fmaxf(m, s[c]);
    red[threadIdx.x] = m; __syncthreads();
    for (int k = 128; k > 0; k >>= 1) {
        if (threadIdx.x < k) red[threadIdx.x] = fmaxf(red[threadIdx.x], red[threadIdx.x + k]);
        __syncthreads();
    }
    m = red[0]; __syncthreads();
    float sum = 0.f;
    for (int c = threadIdx.x; c < cN; c += 256) sum += expf(s[c] - m);
    red[threadIdx.x] = sum; __syncthreads();
    for (int k = 128; k > 0; k >>= 1) {
        if (threadIdx.x < k) red[threadIdx.x] += red[threadIdx.x + k];
        __syncthreads();
    }
    const float inv = 1.f / red[0];
    for (int c = threadIdx.x; c < cN; c += 256)
        p[c] = (half_t)(expf(s[c] - m) * inv);
}

// merge attention heads: O f32 [bh,n,96] -> f16 [T,C]
__global__ void attn_merge_kernel(const float* __restrict__ O, half_t* __restrict__ ao) {
    long long idx = (long long)blockIdx.x * blockDim.x + threadIdx.x;
    if (idx >= (long long)cT * cC) return;
    int c = (int)(idx % cC);
    long long t = idx / cC;
    int n = (int)(t % cN), b = (int)(t / cN);
    int h = c / cHD, d = c % cHD;
    long long bh = (long long)b * cNH + h;
    ao[idx] = (half_t)O[bh * cN * cHDP + (long long)n * cHDP + d];
}

// top-2 gating -> combine weights cw [T,E]
__global__ void gate_kernel(const half_t* __restrict__ xn, const float* __restrict__ wg,
                            float* __restrict__ cw)
{
    int t = blockIdx.x * blockDim.x + threadIdx.x;
    if (t >= cT) return;
    float logit[cE];
    #pragma unroll
    for (int e = 0; e < cE; ++e) logit[e] = 0.f;
    const half_t* x = xn + (long long)t * cC;
    for (int c = 0; c < cC; ++c) {
        float v = (float)x[c];
        #pragma unroll
        for (int e = 0; e < cE; ++e) logit[e] += v * wg[(long long)c * cE + e];
    }
    int i0 = 0;
    #pragma unroll
    for (int e = 1; e < cE; ++e) if (logit[e] > logit[i0]) i0 = e;
    int i1 = -1;
    #pragma unroll
    for (int e = 0; e < cE; ++e)
        if (e != i0 && (i1 < 0 || logit[e] > logit[i1])) i1 = e;
    float e1 = expf(logit[i1] - logit[i0]);
    float inv = 1.f / (1.f + e1);
    #pragma unroll
    for (int e = 0; e < cE; ++e) cw[(long long)t * cE + e] = 0.f;
    cw[(long long)t * cE + i0] = inv;
    cw[(long long)t * cE + i1] = e1 * inv;
}

// out = X1 + g_mlp * moe
__global__ void final_resid_kernel(const float* __restrict__ X1, const float* __restrict__ mod,
                                   const float* __restrict__ moe, float* __restrict__ out)
{
    long long idx = (long long)blockIdx.x * blockDim.x + threadIdx.x;
    if (idx >= (long long)cT * cC) return;
    int c = (int)(idx % cC);
    int b = (int)((idx / cC) / cN);
    out[idx] = X1[idx] + mod[(long long)b * c6C + 5 * cC + c] * moe[idx];
}

// ==========================================================================
extern "C" void kernel_launch(void* const* d_in, const int* in_sizes, int n_in,
                              void* d_out, int out_size, void* d_ws, size_t ws_size,
                              hipStream_t stream)
{
    const float* x       = (const float*)d_in[0];
    const float* c       = (const float*)d_in[1];
    const float* w_adaln = (const float*)d_in[2];
    const float* b_adaln = (const float*)d_in[3];
    const float* w_qkv   = (const float*)d_in[4];
    const float* b_qkv   = (const float*)d_in[5];
    const float* w_proj  = (const float*)d_in[6];
    const float* b_proj  = (const float*)d_in[7];
    const float* w_gate  = (const float*)d_in[8];
    const float* w1      = (const float*)d_in[9];
    const float* b1      = (const float*)d_in[10];
    const float* w2      = (const float*)d_in[11];
    const float* b2      = (const float*)d_in[12];
    float* out = (float*)d_out;

    // bump allocator over workspace
    char* wsb = (char*)d_ws;
    size_t off = 0;
    auto alloc = [&](size_t bytes) -> void* {
        off = (off + 255) & ~(size_t)255;
        void* p = wsb + off;
        off += bytes;
        return p;
    };
    float*  cs      = (float*) alloc((size_t)cB * cC * 4);
    float*  mod     = (float*) alloc((size_t)cB * c6C * 4);
    half_t* h16     = (half_t*)alloc((size_t)cT * cC * 2);
    half_t* wqkv16  = (half_t*)alloc((size_t)cC * c3C * 2);
    half_t* wproj16 = (half_t*)alloc((size_t)cC * cC * 2);
    half_t* w1_16   = (half_t*)alloc((size_t)cE * cC * cH * 2);
    half_t* w2_16   = (half_t*)alloc((size_t)cE * cH * cC * 2);
    half_t* qkv16   = (half_t*)alloc((size_t)cT * c3C * 2);
    half_t* Qp      = (half_t*)alloc((size_t)cBH * cN * cHDP * 2);
    half_t* Kt      = (half_t*)alloc((size_t)cBH * cHDP * cN * 2);
    half_t* Vp      = (half_t*)alloc((size_t)cBH * cN * cHDP * 2);
    float*  scores  = (float*) alloc((size_t)cNH * cN * cN * 4);   // per-b reuse
    half_t* probs   = (half_t*)alloc((size_t)cNH * cN * cN * 2);   // per-b reuse
    float*  O       = (float*) alloc((size_t)cBH * cN * cHDP * 4);
    half_t* ao16    = (half_t*)alloc((size_t)cT * cC * 2);
    float*  X1      = (float*) alloc((size_t)cT * cC * 4);
    half_t* xn16    = (half_t*)alloc((size_t)cT * cC * 2);
    float*  cw      = (float*) alloc((size_t)cT * cE * 4);
    half_t* hmid16  = (half_t*)alloc((size_t)cT * cH * 2);
    float*  moe     = (float*) alloc((size_t)cT * cC * 4);

    auto blocks = [](long long n) { return (unsigned)((n + 255) / 256); };

    // 1) adaLN modulation: mod = SiLU(c) @ w_adaln + b_adaln
    silu_kernel<<<blocks(cB * cC), 256, 0, stream>>>(c, cs, cB * cC);
    adaln_kernel<<<blocks((long long)cB * c6C), 256, 0, stream>>>(cs, w_adaln, b_adaln, mod);

    // 2) weight conversions to f16 (WMMA operands)
    f32_to_f16_kernel<<<blocks((long long)cC * c3C), 256, 0, stream>>>(w_qkv, wqkv16, (long long)cC * c3C);
    f32_to_f16_kernel<<<blocks((long long)cC * cC), 256, 0, stream>>>(w_proj, wproj16, (long long)cC * cC);
    f32_to_f16_kernel<<<blocks((long long)cE * cC * cH), 256, 0, stream>>>(w1, w1_16, (long long)cE * cC * cH);
    f32_to_f16_kernel<<<blocks((long long)cE * cH * cC), 256, 0, stream>>>(w2, w2_16, (long long)cE * cH * cC);

    // 3) LN + modulate (MSA) -> h16
    ln_mod_kernel<<<cT, 256, 0, stream>>>(x, mod, 0, cC, h16);

    // 4) QKV projection: [T,C] x [C,3C] -> qkv16 (f16, +bias)  [FULL + TDM]
    wmma_gemm_kernel<true, true><<<dim3(c3C / 128, cT / 64, 1), 256, 0, stream>>>(
        h16, wqkv16, b_qkv, qkv16,
        cT, c3C, cC, cC, c3C, c3C, 0, 0, 0,
        1.f, 1, nullptr, nullptr, 0, 0, 0);

    // 5) split into padded Q / K^T / V
    qkv_split_kernel<<<blocks((long long)cBH * cN * cHDP), 256, 0, stream>>>(qkv16, Qp, Kt, Vp);

    // 6) attention per batch b (16 heads batched via grid.z); scores buffer reused
    const float isq = 0.1178511301977579f; // 72^-0.5
    for (int b = 0; b < cB; ++b) {
        // scores = (Q x K^T) * isq   [1024,1024] per head, K=96 (zero-padded)
        wmma_gemm_kernel<false, false><<<dim3(cN / 128, cN / 64, cNH), 256, 0, stream>>>(
            Qp + (long long)b * cNH * cN * cHDP,
            Kt + (long long)b * cNH * cHDP * cN,
            nullptr, scores,
            cN, cN, cHDP, cHDP, cN, cN,
            (long long)cN * cHDP, (long long)cHDP * cN, (long long)cN * cN,
            isq, 0, nullptr, nullptr, 0, 0, 0);
        softmax_kernel<<<dim3(cN, cNH), 256, 0, stream>>>(scores, probs);
        // O = probs x V   [1024,96] per head, K=1024
        wmma_gemm_kernel<false, false><<<dim3(1, cN / 64, cNH), 256, 0, stream>>>(
            probs, Vp + (long long)b * cNH * cN * cHDP, nullptr,
            O + (long long)b * cNH * cN * cHDP,
            cN, cHDP, cN, cN, cHDP, cHDP,
            (long long)cN * cN, (long long)cN * cHDP, (long long)cN * cHDP,
            1.f, 0, nullptr, nullptr, 0, 0, 0);
    }

    // 7) merge heads -> ao16, then output projection + gated residual:
    //    X1 = x + g_msa * (ao @ w_proj + b_proj)   [FULL + TDM]
    attn_merge_kernel<<<blocks((long long)cT * cC), 256, 0, stream>>>(O, ao16);
    wmma_gemm_kernel<true, true><<<dim3(cC / 128, cT / 64, 1), 256, 0, stream>>>(
        ao16, wproj16, b_proj, X1,
        cT, cC, cC, cC, cC, cC, 0, 0, 0,
        1.f, 3, x, mod + 2 * cC, (long long)c6C, cN, 0);

    // 8) LN + modulate (MLP) -> xn16 ; top-2 gating -> cw
    ln_mod_kernel<<<cT, 256, 0, stream>>>(X1, mod, 3 * cC, 4 * cC, xn16);
    gate_kernel<<<blocks(cT), 256, 0, stream>>>(xn16, w_gate, cw);

    // 9) MoE: dense per-expert MLP, cw-weighted accumulation into moe  [FULL + TDM]
    for (int e = 0; e < cE; ++e) {
        wmma_gemm_kernel<true, true><<<dim3(cH / 128, cT / 64, 1), 256, 0, stream>>>(
            xn16, w1_16 + (long long)e * cC * cH, b1 + (long long)e * cH, hmid16,
            cT, cH, cC, cC, cH, cH, 0, 0, 0,
            1.f, 2, nullptr, nullptr, 0, 0, 0);
        wmma_gemm_kernel<true, true><<<dim3(cC / 128, cT / 64, 1), 256, 0, stream>>>(
            hmid16, w2_16 + (long long)e * cH * cC, b2 + (long long)e * cC, moe,
            cT, cC, cH, cH, cC, cC, 0, 0, 0,
            1.f, 4, nullptr, cw + e, (long long)cE, 0, (e == 0) ? 1 : 0);
    }

    // 10) out = X1 + g_mlp * moe
    final_resid_kernel<<<blocks((long long)cT * cC), 256, 0, stream>>>(X1, mod, moe, out);
}
